// LSTMModel_37967510896756
// MI455X (gfx1250) — matrix operand order repaired and checked by reading, not compiled, not run on hardware
//
#include <hip/hip_runtime.h>

// CDNA5 / gfx1250 LSTM: recurrent matmul via v_wmma_f32_16x16x32_bf16,
// W_hh held in registers as B-operands, h/gates exchanged through LDS.

typedef __attribute__((ext_vector_type(16))) __bf16 v16bf;
typedef __attribute__((ext_vector_type(8)))  float  v8f;

#define B_TOTAL 2048
#define T_STEPS 512
#define H 50
#define G 200            // 4*H
#define GSTR 210         // gates LDS row stride in dwords (bank-conflict tuned)
#define KP 64            // padded K (hidden) dimension

union BfPack { v16bf v; unsigned short u[16]; };
union AfPack { v16bf v; uint4 q[2]; };

__device__ __forceinline__ unsigned short f2bf(float f) {
  unsigned int u = __float_as_uint(f);
  u += 0x7FFFu + ((u >> 16) & 1u);            // round-to-nearest-even
  return (unsigned short)(u >> 16);
}
__device__ __forceinline__ float bf2f(unsigned short s) {
  return __uint_as_float(((unsigned int)s) << 16);
}
__device__ __forceinline__ float fsigmoid(float x) {
  return __builtin_amdgcn_rcpf(1.0f + __expf(-x));
}
__device__ __forceinline__ float ftanh_fast(float x) {
  float e = __expf(2.0f * x);
  return (e - 1.0f) * __builtin_amdgcn_rcpf(e + 1.0f);
}

__global__ __launch_bounds__(128) void lstm_wmma_kernel(
    const float* __restrict__ x,     const float* __restrict__ W_ih,
    const float* __restrict__ W_hh,  const float* __restrict__ b_ih,
    const float* __restrict__ b_hh,  const float* __restrict__ W_lin,
    const float* __restrict__ b_lin, float* __restrict__ out)
{
  __shared__ float          gates[16 * GSTR];     // 13.4 KB, f32 gate tiles
  __shared__ unsigned short hbuf[16 * KP];        // 2 KB, h in bf16, zero-padded K
  __shared__ float          wih_s[G];
  __shared__ float          wb_s[G];              // b_ih + b_hh
  __shared__ float          red_s[16 * 8];

  const int tid  = threadIdx.x;
  const int wave = tid >> 5;
  const int lane = tid & 31;
  const int lo   = lane & 15;                     // WMMA column / row-in-half
  const int hi   = lane >> 4;                     // WMMA lane-half
  const int b0   = blockIdx.x * 16;               // batch-tile base row

  // N-tile split across 4 waves: 4,3,3,3 of the 13 gate tiles
  const int nstart = (wave == 0) ? 0 : (4 + (wave - 1) * 3);
  const int ntile  = (wave == 0) ? 4 : 3;
  const int nend   = nstart + ntile;

  // ---- one-time LDS init -------------------------------------------------
  for (int g = tid; g < G; g += 128) {
    wih_s[g] = W_ih[g];
    wb_s[g]  = b_ih[g] + b_hh[g];
  }
  for (int i = tid; i < 16 * KP; i += 128) hbuf[i] = 0;   // h0 = 0 (+K pad)

  // ---- W_hh -> bf16 WMMA B-operand registers (wave-private tiles) --------
  // B tile 32x16: lane = column g, lane-half selects K range of 16,
  // vector element e -> K = k2*32 + hi*16 + e  (contiguous k order, same as A).
  v16bf Breg[8];
  #pragma unroll
  for (int nt = 0; nt < 4; ++nt) {
    const int n = nstart + nt;
    #pragma unroll
    for (int k2 = 0; k2 < 2; ++k2) {
      BfPack p;
      #pragma unroll
      for (int e = 0; e < 16; ++e) {
        const int K = k2 * 32 + hi * 16 + e;
        const int g = n * 16 + lo;
        const float w = (n < nend && K < H && g < G) ? W_hh[g * H + K] : 0.0f;
        p.u[e] = f2bf(w);
      }
      Breg[nt * 2 + k2] = p.v;
    }
  }

  // ---- per-thread cell state: row r, 7 hidden cols starting at jbase -----
  const int r     = tid & 15;
  const int grp   = tid >> 4;
  const int jbase = grp * 7;                       // 8 groups * 7 >= 50
  float creg[7];
  #pragma unroll
  for (int i = 0; i < 7; ++i) creg[i] = 0.0f;

  const float* xrow   = x + (size_t)(b0 + r) * T_STEPS;
  const char*  hbytes = (const char*)hbuf;

  __syncthreads();

  // ======================= time loop (serial) =============================
  for (int t = 0; t < T_STEPS; ++t) {
    // A operands (h tile, 16x64 bf16) in contiguous k order (as B):
    // lane row = lo; q[0]=K(hi*16 .. +7) block, q[1]=K(hi*16+8 .. +15) block.
    AfPack A0, A1;
    {
      const char* base = hbytes + lo * (KP * 2);
      const int   hb   = hi * 32;
      A0.q[0] = *(const uint4*)(base + hb);
      A0.q[1] = *(const uint4*)(base + 16 + hb);
      A1.q[0] = *(const uint4*)(base + 64 + hb);
      A1.q[1] = *(const uint4*)(base + 80 + hb);
    }

    // gates_tile = h @ W_hh^T  (two K-steps of 32, bf16 -> f32 accum)
    #pragma unroll
    for (int nt = 0; nt < 4; ++nt) {
      const int n = nstart + nt;
      if (n < nend) {                               // wave-uniform branch
        v8f acc = {};
        acc = __builtin_amdgcn_wmma_f32_16x16x32_bf16(
                  false, A0.v, false, Breg[nt * 2 + 0], (short)0, acc, false, false);
        acc = __builtin_amdgcn_wmma_f32_16x16x32_bf16(
                  false, A1.v, false, Breg[nt * 2 + 1], (short)0, acc, false, false);
        // D layout: VGPR rr -> row rr + 8*hi, col n*16 + lo
        float* gdst = &gates[(hi * 8) * GSTR + n * 16 + lo];
        #pragma unroll
        for (int rr = 0; rr < 8; ++rr) gdst[rr * GSTR] = acc[rr];
      }
    }
    __syncthreads();

    // elementwise gate update (f32): gates += x*W_ih + (b_ih+b_hh), act, cell
    const float xv = xrow[t];
    if (t + 16 < T_STEPS) __builtin_prefetch(xrow + t + 16, 0, 0);
    const float* grow = &gates[r * GSTR];
    #pragma unroll
    for (int i = 0; i < 7; ++i) {
      const int j = jbase + i;
      if (j < H) {
        float gi = grow[j        ] + xv * wih_s[j        ] + wb_s[j        ];
        float gf = grow[H   + j  ] + xv * wih_s[H   + j  ] + wb_s[H   + j  ];
        float gg = grow[2*H + j  ] + xv * wih_s[2*H + j  ] + wb_s[2*H + j  ];
        float go = grow[3*H + j  ] + xv * wih_s[3*H + j  ] + wb_s[3*H + j  ];
        const float ig = fsigmoid(gi);
        const float fg = fsigmoid(gf);
        const float cg = ftanh_fast(gg);
        const float og = fsigmoid(go);
        const float c  = fg * creg[i] + ig * cg;
        creg[i] = c;
        const float h = og * ftanh_fast(c);
        hbuf[r * KP + j] = f2bf(h);
      }
    }
    __syncthreads();
  }

  // ---- final linear: out[b] = h_T . W_lin + b_lin ------------------------
  float po = 0.0f;
  #pragma unroll
  for (int i = 0; i < 7; ++i) {
    const int j = jbase + i;
    if (j < H) po += bf2f(hbuf[r * KP + j]) * W_lin[j];
  }
  red_s[r * 8 + grp] = po;
  __syncthreads();
  if (tid < 16) {
    float s = b_lin[0];
    #pragma unroll
    for (int g2 = 0; g2 < 8; ++g2) s += red_s[tid * 8 + g2];
    out[b0 + tid] = s;
  }
}

extern "C" void kernel_launch(void* const* d_in, const int* in_sizes, int n_in,
                              void* d_out, int out_size, void* d_ws, size_t ws_size,
                              hipStream_t stream) {
  (void)in_sizes; (void)n_in; (void)d_ws; (void)ws_size; (void)out_size;
  const float* x     = (const float*)d_in[0];
  const float* W_ih  = (const float*)d_in[1];
  const float* W_hh  = (const float*)d_in[2];
  const float* b_ih  = (const float*)d_in[3];
  const float* b_hh  = (const float*)d_in[4];
  const float* W_lin = (const float*)d_in[5];
  const float* b_lin = (const float*)d_in[6];
  float* out = (float*)d_out;

  dim3 grid(B_TOTAL / 16);   // 128 workgroups, one 16-row batch tile each
  dim3 block(128);           // 4 waves: N-tiles split 4/3/3/3
  hipLaunchKernelGGL(lstm_wmma_kernel, grid, block, 0, stream,
                     x, W_ih, W_hh, b_ih, b_hh, W_lin, b_lin, out);
}
